// PointDiffuse_56710748176538
// MI455X (gfx1250) — matrix operand order repaired and checked by reading, not compiled, not run on hardware
//
#include <hip/hip_runtime.h>
#include <hip/hip_bf16.h>

#define BB    4
#define NN    8192
#define CIN   64
#define KKNN  16
#define COUT  128
#define MTOT  (BB*NN*KKNN)      // 524288 rows
#define NKROW (NN*KKNN)         // 131072 rows per batch
#define EPSBN 1e-5f

typedef _Float16 v16h __attribute__((ext_vector_type(16)));
typedef _Float16 h8   __attribute__((ext_vector_type(8)));
typedef float    v8f  __attribute__((ext_vector_type(8)));

// ---------------------------------------------------------------- helpers
__device__ __forceinline__ v16h load_a_frag(const _Float16* ap) {
    // A 16x32 f16 frag: lane holds row M=lane&15; K-halfs in documented order:
    // lane<16: K = {kb..kb+7, kb+16..kb+23}; lane>=16: +8. Caller passes ap
    // already offset by (row*stride + kb + ((lane>=16)?8:0)).
    h8 lo = *(const h8*)(ap);
    h8 hi = *(const h8*)(ap + 16);
    v16h a;
#pragma unroll
    for (int j = 0; j < 8; ++j) { a[j] = lo[j]; a[j + 8] = hi[j]; }
    return a;
}

// ---------------------------------------------------------------- K0: mean xyz
__global__ void k_mean_xyz(const float* __restrict__ pos, float* __restrict__ xyz) {
    int p = blockIdx.x * 256 + threadIdx.x;             // 0..32767
    const float* s = pos + (size_t)p * (KKNN * 3);
    float sx = 0.f, sy = 0.f, sz = 0.f;
#pragma unroll
    for (int k = 0; k < KKNN; ++k) {
        sx += s[k * 3 + 0]; sy += s[k * 3 + 1]; sz += s[k * 3 + 2];
    }
    const float r = 1.0f / (float)KKNN;
    xyz[p * 3 + 0] = sx * r; xyz[p * 3 + 1] = sy * r; xyz[p * 3 + 2] = sz * r;
}

// ---------------------------------------------------------------- K1: KNN (k=16)
__global__ __launch_bounds__(256) void k_knn(const float* __restrict__ xyz,
                                             int* __restrict__ idx) {
    __shared__ float sx[256], sy[256], sz[256];
    const int batch = blockIdx.x >> 5;                  // 32 blocks per batch
    const int q = ((blockIdx.x & 31) << 8) + threadIdx.x;
    const float* base = xyz + (size_t)batch * NN * 3;
    const float qx = base[q * 3 + 0], qy = base[q * 3 + 1], qz = base[q * 3 + 2];

    float bd[16]; int bi[16];
#pragma unroll
    for (int i = 0; i < 16; ++i) { bd[i] = 3.4e38f; bi[i] = 0; }
    float worst = 3.4e38f; int wslot = 0;

    for (int tile = 0; tile < NN / 256; ++tile) {
        const int c = tile * 256 + threadIdx.x;
        sx[threadIdx.x] = base[c * 3 + 0];
        sy[threadIdx.x] = base[c * 3 + 1];
        sz[threadIdx.x] = base[c * 3 + 2];
        __syncthreads();
        for (int j = 0; j < 256; ++j) {
            const float dx = qx - sx[j], dy = qy - sy[j], dz = qz - sz[j];
            const float d = dx * dx + dy * dy + dz * dz;
            if (d < worst) {
                const int cand = tile * 256 + j;
#pragma unroll
                for (int i = 0; i < 16; ++i)
                    if (i == wslot) { bd[i] = d; bi[i] = cand; }
                worst = bd[0]; wslot = 0;
#pragma unroll
                for (int i = 1; i < 16; ++i)
                    if (bd[i] > worst) { worst = bd[i]; wslot = i; }
            }
        }
        __syncthreads();
    }
    int* op = idx + ((size_t)batch * NN + q) * KKNN;
#pragma unroll
    for (int i = 0; i < 16; ++i) op[i] = bi[i];
}

// ------------------------------------------- K2: weights->f16 + zero stats
__global__ void k_prep(const float* __restrict__ W1, const float* __restrict__ W2,
                       const float* __restrict__ W3, _Float16* __restrict__ W1h,
                       _Float16* __restrict__ W2h, _Float16* __restrict__ W3h,
                       float* __restrict__ stats) {
    const int t = blockIdx.x * 256 + threadIdx.x;
    if (t < 512) stats[t] = 0.f;                        // sum1,sq1,sum2,sq2
    if (t < 96 * 128) {                                 // W1 padded 67 -> 96 rows
        const int k = t >> 7, n = t & 127;
        W1h[t] = (k < CIN + 3) ? (_Float16)W1[k * 128 + n] : (_Float16)0.f;
    } else if (t < 96 * 128 + 128 * 128) {
        const int i = t - 96 * 128;
        W2h[i] = (_Float16)W2[i];
    } else if (t < 96 * 128 + 2 * 128 * 128) {
        const int i = t - 96 * 128 - 128 * 128;
        W3h[i] = (_Float16)W3[i];
    }
}

// ---------------------------------------------------------------- GEMM1
// rows = gather(features, idx) ++ pos, K-dim 67 padded to 96. Out: X1 f16 + stats.
__global__ __launch_bounds__(256) void k_gemm1(
    const float* __restrict__ feats, const float* __restrict__ pos,
    const int* __restrict__ idx, const _Float16* __restrict__ Wh,
    _Float16* __restrict__ X1, float* __restrict__ g_sum, float* __restrict__ g_sq) {
    constexpr int AST = 104;                            // 96 + 8 pad (mult of 8)
    __shared__ __align__(16) _Float16 As[128 * AST];
    __shared__ float s_sum[128], s_sq[128];
    const int t = threadIdx.x;
    if (t < 128) { s_sum[t] = 0.f; s_sq[t] = 0.f; }
    const size_t rowBase = (size_t)blockIdx.x * 128;

    { // ---- A tile: gather + concat + f16 convert (2 threads / row)
        const int r = t >> 1, h = t & 1;
        const size_t g = rowBase + r;
        const int b = (int)(g >> 17);                   // / 131072
        const int nb = idx[g];
        const float* fr = feats + ((size_t)b * NN + nb) * CIN + h * 32;
        _Float16* dst = As + r * AST + h * 32;
#pragma unroll
        for (int i = 0; i < 8; ++i) {
            const float4 v = *(const float4*)(fr + i * 4);
            h8 o; o[0] = (_Float16)v.x; o[1] = (_Float16)v.y;
            o[2] = (_Float16)v.z; o[3] = (_Float16)v.w;
            // store 4 halfs (low half of an h8 write would be wrong) -> 8B store
            *(__attribute__((ext_vector_type(4))) _Float16*)(dst + i * 4) =
                *(__attribute__((ext_vector_type(4))) _Float16*)&o;
        }
        if (h == 1) {
            const float* pp = pos + g * 3;
            h8 z = (h8)0;
            h8 pv = (h8)0;
            pv[0] = (_Float16)pp[0]; pv[1] = (_Float16)pp[1]; pv[2] = (_Float16)pp[2];
            _Float16* d2 = As + r * AST;
            *(h8*)(d2 + 64) = pv;                       // cols 64..71 (3 pos + zeros)
            *(h8*)(d2 + 72) = z; *(h8*)(d2 + 80) = z; *(h8*)(d2 + 88) = z;
        }
    }
    __syncthreads();

    const int wave = t >> 5, lane = t & 31;
    const int mrow = wave * 16 + (lane & 15);
    const int khalf = (lane >> 4) ? 8 : 0;
    v8f acc[8];
    const v8f vz = (v8f)0.f;
#pragma unroll
    for (int nt = 0; nt < 8; ++nt) acc[nt] = vz;

#pragma unroll
    for (int ks = 0; ks < 3; ++ks) {                    // K = 96
        const int kb = ks * 32;
        const v16h a = load_a_frag(As + mrow * AST + kb + khalf);
        const _Float16* wrow = Wh + (size_t)(kb + khalf) * 128 + (lane & 15);
#pragma unroll
        for (int nt = 0; nt < 8; ++nt) {
            v16h bfr;
            for (int s = 0; s < 16; ++s) bfr[s] = wrow[(size_t)((s & 7) + (s >> 3) * 16) * 128 + nt * 16];
            acc[nt] = __builtin_amdgcn_wmma_f32_16x16x32_f16(
                false, a, false, bfr, (short)0, acc[nt], false, false);
        }
    }

    // ---- epilogue: store f16 + column stats
    const int colLo = lane & 15;
    const int rhi = (lane >> 4) * 8;
#pragma unroll
    for (int nt = 0; nt < 8; ++nt) {
        float sum = 0.f, sq = 0.f;
#pragma unroll
        for (int j = 0; j < 8; ++j) {
            const float v = acc[nt][j];
            const size_t row = rowBase + wave * 16 + j + rhi;
            X1[row * 128 + nt * 16 + colLo] = (_Float16)v;
            sum += v; sq += v * v;
        }
        sum += __shfl_xor(sum, 16, 32);
        sq  += __shfl_xor(sq, 16, 32);
        if (lane < 16) {
            atomicAdd(&s_sum[nt * 16 + colLo], sum);
            atomicAdd(&s_sq[nt * 16 + colLo], sq);
        }
    }
    __syncthreads();
    if (t < 128) {
        unsafeAtomicAdd(&g_sum[t], s_sum[t]);
        unsafeAtomicAdd(&g_sq[t], s_sq[t]);
    }
}

// ------------------------------------------- finalize BN: fold to scale/shift
__global__ void k_finalize(const float* __restrict__ sum, const float* __restrict__ sq,
                           const float* __restrict__ gamma, const float* __restrict__ beta,
                           float* __restrict__ scale, float* __restrict__ shift) {
    const int c = threadIdx.x;
    const float inv_m = 1.0f / (float)MTOT;
    const float mean = sum[c] * inv_m;
    const float var = sq[c] * inv_m - mean * mean;
    const float sc = gamma[c] * rsqrtf(var + EPSBN);
    scale[c] = sc;
    shift[c] = beta[c] - mean * sc;
}

// ---------------------------------------------------------------- GEMM2
// X2 = (relu(bn(X1))) @ W2 ; stats for BN2.
__global__ __launch_bounds__(256) void k_gemm2(
    const _Float16* __restrict__ Xin, const _Float16* __restrict__ Wh,
    const float* __restrict__ scale, const float* __restrict__ shift,
    _Float16* __restrict__ Xout, float* __restrict__ g_sum, float* __restrict__ g_sq) {
    constexpr int AST = 136;                            // 128 + 8 pad
    __shared__ __align__(16) _Float16 As[128 * AST];
    __shared__ float s_scale[128], s_shift[128], s_sum[128], s_sq[128];
    const int t = threadIdx.x;
    if (t < 128) { s_scale[t] = scale[t]; s_shift[t] = shift[t];
                   s_sum[t] = 0.f; s_sq[t] = 0.f; }
    __syncthreads();
    const size_t rowBase = (size_t)blockIdx.x * 128;

    { // ---- A tile: BN + ReLU applied on load (2 threads / row)
        const int r = t >> 1, h = t & 1;
        const _Float16* src = Xin + (rowBase + r) * 128 + h * 64;
        _Float16* dst = As + r * AST + h * 64;
#pragma unroll
        for (int i = 0; i < 8; ++i) {
            const h8 v = *(const h8*)(src + i * 8);
            h8 o;
#pragma unroll
            for (int j = 0; j < 8; ++j) {
                const int c = h * 64 + i * 8 + j;
                const float f = (float)v[j] * s_scale[c] + s_shift[c];
                o[j] = (_Float16)fmaxf(f, 0.f);
            }
            *(h8*)(dst + i * 8) = o;
        }
    }
    __syncthreads();

    const int wave = t >> 5, lane = t & 31;
    const int mrow = wave * 16 + (lane & 15);
    const int khalf = (lane >> 4) ? 8 : 0;
    v8f acc[8];
    const v8f vz = (v8f)0.f;
#pragma unroll
    for (int nt = 0; nt < 8; ++nt) acc[nt] = vz;

#pragma unroll
    for (int ks = 0; ks < 4; ++ks) {                    // K = 128
        const int kb = ks * 32;
        const v16h a = load_a_frag(As + mrow * AST + kb + khalf);
        const _Float16* wrow = Wh + (size_t)(kb + khalf) * 128 + (lane & 15);
#pragma unroll
        for (int nt = 0; nt < 8; ++nt) {
            v16h bfr;
            for (int s = 0; s < 16; ++s) bfr[s] = wrow[(size_t)((s & 7) + (s >> 3) * 16) * 128 + nt * 16];
            acc[nt] = __builtin_amdgcn_wmma_f32_16x16x32_f16(
                false, a, false, bfr, (short)0, acc[nt], false, false);
        }
    }

    const int colLo = lane & 15;
    const int rhi = (lane >> 4) * 8;
#pragma unroll
    for (int nt = 0; nt < 8; ++nt) {
        float sum = 0.f, sq = 0.f;
#pragma unroll
        for (int j = 0; j < 8; ++j) {
            const float v = acc[nt][j];
            const size_t row = rowBase + wave * 16 + j + rhi;
            Xout[row * 128 + nt * 16 + colLo] = (_Float16)v;
            sum += v; sq += v * v;
        }
        sum += __shfl_xor(sum, 16, 32);
        sq  += __shfl_xor(sq, 16, 32);
        if (lane < 16) {
            atomicAdd(&s_sum[nt * 16 + colLo], sum);
            atomicAdd(&s_sq[nt * 16 + colLo], sq);
        }
    }
    __syncthreads();
    if (t < 128) {
        unsafeAtomicAdd(&g_sum[t], s_sum[t]);
        unsafeAtomicAdd(&g_sq[t], s_sq[t]);
    }
}

// ---------------------------------------------------------------- GEMM3 + maxpool
// out[p,:] = max_k( relu(bn2(X2)) @ W3 ) + b3 ; wave w owns one point's 16 rows.
__global__ __launch_bounds__(256) void k_gemm3(
    const _Float16* __restrict__ Xin, const _Float16* __restrict__ Wh,
    const float* __restrict__ scale, const float* __restrict__ shift,
    const float* __restrict__ b3, float* __restrict__ out) {
    constexpr int AST = 136;
    __shared__ __align__(16) _Float16 As[128 * AST];
    __shared__ float s_scale[128], s_shift[128];
    const int t = threadIdx.x;
    if (t < 128) { s_scale[t] = scale[t]; s_shift[t] = shift[t]; }
    __syncthreads();
    const size_t rowBase = (size_t)blockIdx.x * 128;

    {
        const int r = t >> 1, h = t & 1;
        const _Float16* src = Xin + (rowBase + r) * 128 + h * 64;
        _Float16* dst = As + r * AST + h * 64;
#pragma unroll
        for (int i = 0; i < 8; ++i) {
            const h8 v = *(const h8*)(src + i * 8);
            h8 o;
#pragma unroll
            for (int j = 0; j < 8; ++j) {
                const int c = h * 64 + i * 8 + j;
                const float f = (float)v[j] * s_scale[c] + s_shift[c];
                o[j] = (_Float16)fmaxf(f, 0.f);
            }
            *(h8*)(dst + i * 8) = o;
        }
    }
    __syncthreads();

    const int wave = t >> 5, lane = t & 31;
    const int mrow = wave * 16 + (lane & 15);
    const int khalf = (lane >> 4) ? 8 : 0;
    v8f acc[8];
    const v8f vz = (v8f)0.f;
#pragma unroll
    for (int nt = 0; nt < 8; ++nt) acc[nt] = vz;

#pragma unroll
    for (int ks = 0; ks < 4; ++ks) {
        const int kb = ks * 32;
        const v16h a = load_a_frag(As + mrow * AST + kb + khalf);
        const _Float16* wrow = Wh + (size_t)(kb + khalf) * 128 + (lane & 15);
#pragma unroll
        for (int nt = 0; nt < 8; ++nt) {
            v16h bfr;
            for (int s = 0; s < 16; ++s) bfr[s] = wrow[(size_t)((s & 7) + (s >> 3) * 16) * 128 + nt * 16];
            acc[nt] = __builtin_amdgcn_wmma_f32_16x16x32_f16(
                false, a, false, bfr, (short)0, acc[nt], false, false);
        }
    }

    // ---- max over the wave's 16 rows (= one point's K neighbors)
    const int colLo = lane & 15;
    const size_t point = (size_t)blockIdx.x * 8 + wave;
#pragma unroll
    for (int nt = 0; nt < 8; ++nt) {
        float m = acc[nt][0];
#pragma unroll
        for (int j = 1; j < 8; ++j) m = fmaxf(m, acc[nt][j]);
        m = fmaxf(m, __shfl_xor(m, 16, 32));
        if (lane < 16) {
            const int c = nt * 16 + colLo;
            out[point * 128 + c] = m + b3[c];
        }
    }
}

// ---------------------------------------------------------------- launch
extern "C" void kernel_launch(void* const* d_in, const int* in_sizes, int n_in,
                              void* d_out, int out_size, void* d_ws, size_t ws_size,
                              hipStream_t stream) {
    const float* feats = (const float*)d_in[0];
    const float* pos   = (const float*)d_in[1];
    const float* W1    = (const float*)d_in[2];
    const float* g1    = (const float*)d_in[4];
    const float* bt1   = (const float*)d_in[5];
    const float* W2    = (const float*)d_in[6];
    const float* g2    = (const float*)d_in[8];
    const float* bt2   = (const float*)d_in[9];
    const float* W3    = (const float*)d_in[10];
    const float* b3    = (const float*)d_in[11];
    float* out = (float*)d_out;

    char* ws = (char*)d_ws;
    size_t off = 0;
    auto take = [&](size_t bytes) { char* p = ws + off; off = (off + bytes + 255) & ~(size_t)255; return p; };
    float*    xyz  = (float*)   take((size_t)BB * NN * 3 * 4);
    int*      idxb = (int*)     take((size_t)MTOT * 4);
    _Float16* W1h  = (_Float16*)take((size_t)96 * 128 * 2);
    _Float16* W2h  = (_Float16*)take((size_t)128 * 128 * 2);
    _Float16* W3h  = (_Float16*)take((size_t)128 * 128 * 2);
    float*    st   = (float*)   take((size_t)1024 * 4);
    _Float16* X1   = (_Float16*)take((size_t)MTOT * 128 * 2);
    _Float16* X2   = (_Float16*)take((size_t)MTOT * 128 * 2);
    float* sum1 = st, *sq1 = st + 128, *sum2 = st + 256, *sq2 = st + 384;
    float* sc1 = st + 512, *sh1 = st + 640, *sc2 = st + 768, *sh2 = st + 896;

    k_mean_xyz<<<(BB * NN) / 256, 256, 0, stream>>>(pos, xyz);
    k_knn<<<(BB * NN) / 256, 256, 0, stream>>>(xyz, idxb);
    k_prep<<<(96 * 128 + 2 * 128 * 128 + 255) / 256, 256, 0, stream>>>(
        W1, W2, W3, W1h, W2h, W3h, st);
    k_gemm1<<<MTOT / 128, 256, 0, stream>>>(feats, pos, idxb, W1h, X1, sum1, sq1);
    k_finalize<<<1, 128, 0, stream>>>(sum1, sq1, g1, bt1, sc1, sh1);
    k_gemm2<<<MTOT / 128, 256, 0, stream>>>(X1, W2h, sc1, sh1, X2, sum2, sq2);
    k_finalize<<<1, 128, 0, stream>>>(sum2, sq2, g2, bt2, sc2, sh2);
    k_gemm3<<<MTOT / 128, 256, 0, stream>>>(X2, W3h, sc2, sh2, b3, out);
}
